// MultiHeadAttention_46591805227455
// MI455X (gfx1250) — compile-verified, matrix operand order checked by reading, not run on hardware
//
#include <hip/hip_runtime.h>
#include <hip/hip_bf16.h>

typedef __attribute__((ext_vector_type(16))) _Float16 v16h;
typedef __attribute__((ext_vector_type(8)))  _Float16 v8h;
typedef __attribute__((ext_vector_type(8)))  float    v8f;

#define B_   2
#define S_   2048
#define DIN  1024
#define DOUT 1024
#define H_   16
#define HD_  64

// ---------------------------------------------------------------------------
// helpers
// ---------------------------------------------------------------------------
__device__ __forceinline__ v8f zero8() {
  v8f z = {0.f, 0.f, 0.f, 0.f, 0.f, 0.f, 0.f, 0.f};
  return z;
}

// Load one 16-wide f16 operand row-slice per the CDNA5 16-bit A/B packing:
// a lane's 16 halves are two contiguous 8-half chunks: [0,8) and [16,24)
// relative to the lane's base offset (caller adds base = (lane&16)?8:0).
__device__ __forceinline__ v16h load_op16(const _Float16* p) {
  v8h lo = *(const v8h*)(p);
  v8h hi = *(const v8h*)(p + 16);
  v16h r;
#pragma unroll
  for (int i = 0; i < 8; ++i) { r[i] = lo[i]; r[i + 8] = hi[i]; }
  return r;
}

__device__ __forceinline__ v8f wmma_f16(v16h a, v16h b, v8f c) {
  // (neg_a, A, neg_b, B, c_mod, C, reuse_a, reuse_b)
  return __builtin_amdgcn_wmma_f32_16x16x32_f16(false, a, false, b, (short)0, c,
                                                false, false);
}

// Async DMA: copy 4KB (contiguous) from global to LDS, 8 x (32 lanes x 16B).
// Per ISA 15.18.3: each lane copies 16B from its global address to its LDS address;
// the instruction immediate offset advances BOTH addresses. Tracked by ASYNCcnt.
__device__ __forceinline__ void async_copy_4k(unsigned lds, const char* g) {
#define MHA_ASYNC_LD(OFF)                                                     \
  asm volatile("global_load_async_to_lds_b128 %0, %1, off offset:" #OFF       \
               :: "v"(lds), "v"(g) : "memory")
  MHA_ASYNC_LD(0);    MHA_ASYNC_LD(512);  MHA_ASYNC_LD(1024); MHA_ASYNC_LD(1536);
  MHA_ASYNC_LD(2048); MHA_ASYNC_LD(2560); MHA_ASYNC_LD(3072); MHA_ASYNC_LD(3584);
#undef MHA_ASYNC_LD
}

__device__ __forceinline__ void wait_async0() {
  asm volatile("s_wait_asynccnt 0x0" ::: "memory");
}

// ---------------------------------------------------------------------------
// Kernel 1: f32 -> f16 pack; weights transposed so B-operand is K-contiguous
// ---------------------------------------------------------------------------
__global__ void __launch_bounds__(256)
mha_convert(const float* __restrict__ x,
            const float* __restrict__ Wq, const float* __restrict__ Wk,
            const float* __restrict__ Wv, const float* __restrict__ Wo,
            _Float16* __restrict__ xh,
            _Float16* __restrict__ WqT, _Float16* __restrict__ WkT,
            _Float16* __restrict__ WvT, _Float16* __restrict__ WoT) {
  int tid = blockIdx.x * blockDim.x + threadIdx.x;
  int stride = gridDim.x * blockDim.x;
  const int NX = B_ * S_ * DIN;
  for (int i = tid; i < NX; i += stride) xh[i] = (_Float16)x[i];
  const int NW = DIN * DOUT;
  for (int i = tid; i < NW; i += stride) {
    int k = i >> 10, n = i & 1023;      // W is (K=DIN, N=DOUT) row-major
    int t = n * DIN + k;                // W^T is (N, K) row-major
    WqT[t] = (_Float16)Wq[i];
    WkT[t] = (_Float16)Wk[i];
    WvT[t] = (_Float16)Wv[i];
    WoT[t] = (_Float16)Wo[i];
  }
}

// ---------------------------------------------------------------------------
// GEMM epilogue: mode 0/1 = RoPE + write (b,h,s,hd) f16 (Q/K)
//                mode 2   = write V transposed (b,h,hd,s) f16
//                mode 3   = write f32 row-major (output projection)
// C tile layout: VGPR r, lane L -> M = r + 8*(L/16), N = L%16
// ---------------------------------------------------------------------------
__device__ __forceinline__ void
mha_epilogue_tile(v8f c, int mt, int nt, int mode, void* dst, int hg, int cl) {
#pragma unroll
  for (int r = 0; r < 8; ++r) {
    float val = c[r];
    float partner = __shfl_xor(val, 1, 32);   // RoPE pair lives in lane^1
    int m = mt + r + 8 * hg;
    int n = nt + cl;
    if (mode <= 1) {
      int srow = m & (S_ - 1);
      int bb = m >> 11;                       // S_ == 2048
      int hh = n >> 6;
      int d = n & 63;
      int pp = d >> 1;
      // inv_freq = 10000^(-p/32) = exp(-p * ln(10000)/32)
      float ang = (float)srow * __expf(-(float)pp * 0.28782313662425572f);
      float sv, cv;
      __sincosf(ang, &sv, &cv);
      float out = (d & 1) ? (partner * sv + val * cv)
                          : (val * cv - partner * sv);
      _Float16* q = (_Float16*)dst;
      q[(((size_t)bb * H_ + hh) * S_ + srow) * HD_ + d] = (_Float16)out;
    } else if (mode == 2) {
      int srow = m & (S_ - 1);
      int bb = m >> 11;
      int hh = n >> 6;
      int d = n & 63;
      _Float16* v = (_Float16*)dst;
      v[(((size_t)bb * H_ + hh) * HD_ + d) * S_ + srow] = (_Float16)val;
    } else {
      float* o = (float*)dst;
      o[(size_t)m * DOUT + n] = val;
    }
  }
}

// ---------------------------------------------------------------------------
// Kernel 2/4: C = A(MxK) * B(KxN), BT = (NxK) row-major, f16 WMMA, f32 acc.
// 8 waves/block, each wave a 32x64 tile (8 WMMA / 12 b128 loads per k-step);
// block covers 64x256.
// ---------------------------------------------------------------------------
__global__ void __launch_bounds__(256)
mha_gemm_wmma(const _Float16* __restrict__ A, const _Float16* __restrict__ BT,
              void* __restrict__ dst, int M, int N, int K, int mode) {
  const int lane = threadIdx.x & 31;
  const int wave = threadIdx.x >> 5;
  const int hg = lane >> 4;
  const int cl = lane & 15;
  const int kbase = (lane & 16) ? 8 : 0;

  const int m0 = blockIdx.x * 64 + (wave & 1) * 32;
  const int n0 = blockIdx.y * 256 + (wave >> 1) * 64;
  (void)M; (void)N;

  v8f acc[2][4];
#pragma unroll
  for (int mi = 0; mi < 2; ++mi)
#pragma unroll
    for (int ni = 0; ni < 4; ++ni) acc[mi][ni] = zero8();

  const _Float16* arow[2];
  const _Float16* brow[4];
#pragma unroll
  for (int mi = 0; mi < 2; ++mi)
    arow[mi] = A + (size_t)(m0 + mi * 16 + cl) * K;
#pragma unroll
  for (int ni = 0; ni < 4; ++ni)
    brow[ni] = BT + (size_t)(n0 + ni * 16 + cl) * K;

  for (int kk = 0; kk < K; kk += 32) {
    const int ko = kk + kbase;
    v16h a[2], b[4];
#pragma unroll
    for (int mi = 0; mi < 2; ++mi) a[mi] = load_op16(arow[mi] + ko);
#pragma unroll
    for (int ni = 0; ni < 4; ++ni) b[ni] = load_op16(brow[ni] + ko);
    __builtin_prefetch(arow[0] + ko + 128, 0, 3);   // global_prefetch_b8
    __builtin_prefetch(brow[0] + ko + 128, 0, 3);
#pragma unroll
    for (int mi = 0; mi < 2; ++mi)
#pragma unroll
      for (int ni = 0; ni < 4; ++ni)
        acc[mi][ni] = wmma_f16(a[mi], b[ni], acc[mi][ni]);
  }

#pragma unroll
  for (int mi = 0; mi < 2; ++mi)
#pragma unroll
    for (int ni = 0; ni < 4; ++ni)
      mha_epilogue_tile(acc[mi][ni], m0 + mi * 16, n0 + ni * 16, mode, dst, hg, cl);
}

// ---------------------------------------------------------------------------
// Kernel 3: causal flash attention. Block = (b, h, 64 q-rows); 4 waves x 16 q.
// K tiles arrive via double-buffered GLOBAL_LOAD_ASYNC_TO_LDS_B128 (ASYNCcnt);
// per 32-key chunk: 4 WMMA for S = Q*K^T (B-operand from LDS), online softmax
// with shfl_xor half-wave reductions, P through LDS (C->A layout transpose),
// 4 WMMA for PV (V from global, transposed layout). Q pre-scaled by 1/8.
// ---------------------------------------------------------------------------
__global__ void __launch_bounds__(128)
mha_flash_wmma(const _Float16* __restrict__ qh, const _Float16* __restrict__ kh,
               const _Float16* __restrict__ vT, _Float16* __restrict__ ctx) {
  __shared__ __align__(16) _Float16 Klds[4][2][32 * HD_];  // 4KB x2 per wave
  __shared__ __align__(16) _Float16 Plds[4][16][40];       // 80B pitch (16B-aligned)

  const int lane = threadIdx.x & 31;
  const int wave = threadIdx.x >> 5;
  const int hg = lane >> 4;
  const int cl = lane & 15;
  const int kbase = (lane & 16) ? 8 : 0;

  const int bh = blockIdx.x >> 5;   // b*H + h  (S/64 == 32 q-tiles)
  const int qt = blockIdx.x & 31;
  const int q0 = qt * 64 + wave * 16;

  // Q A-operand: 16 rows x 64 -> two 16x32 tiles, scaled by 1/sqrt(HD)
  const _Float16* qrowp = qh + ((size_t)bh * S_ + q0 + cl) * HD_;
  v16h aq0 = load_op16(qrowp + kbase);
  v16h aq1 = load_op16(qrowp + 32 + kbase);
#pragma unroll
  for (int i = 0; i < 16; ++i) {
    aq0[i] = aq0[i] * (_Float16)0.125f;
    aq1[i] = aq1[i] * (_Float16)0.125f;
  }

  v8f o0 = zero8(), o1 = zero8(), o2 = zero8(), o3 = zero8();
  float mrun[8], lrun[8];
#pragma unroll
  for (int r = 0; r < 8; ++r) { mrun[r] = -1e30f; lrun[r] = 0.f; }

  _Float16* Pw = &Plds[wave][0][0];
  const _Float16* kchunk_base = kh + (size_t)bh * S_ * HD_;   // 4KB per 32 keys
  const unsigned lane16 = (unsigned)lane * 16u;

  const int nChunks = (q0 + 16 + 31) >> 5;   // keys [0, q0+16) needed

  // prologue: async-stage chunk 0 into buffer 0
  async_copy_4k((unsigned)(size_t)(void*)&Klds[wave][0][0] + lane16,
                (const char*)kchunk_base + lane16);

  for (int j = 0; j < nChunks; ++j) {
    const int k0 = j * 32;
    const _Float16* kbuf = &Klds[wave][j & 1][0];

    wait_async0();                       // chunk j resident in LDS

    // ---- pull B-operand K tiles out of LDS (ds_load_b128 x2 each) ----
    v16h bk00 = load_op16(kbuf + (size_t)cl * HD_ + kbase);         // keys k0..+15, hd 0..31
    v16h bk01 = load_op16(kbuf + (size_t)cl * HD_ + 32 + kbase);    // keys k0..+15, hd 32..63
    v16h bk10 = load_op16(kbuf + (size_t)(16 + cl) * HD_ + kbase);  // keys +16..31
    v16h bk11 = load_op16(kbuf + (size_t)(16 + cl) * HD_ + 32 + kbase);

    // ---- overlap: async-stage chunk j+1 while we compute chunk j ----
    if (j + 1 < nChunks) {
      async_copy_4k((unsigned)(size_t)(void*)&Klds[wave][(j + 1) & 1][0] + lane16,
                    (const char*)(kchunk_base + (size_t)(k0 + 32) * HD_) + lane16);
    }

    // ---- scores for 32 keys: two 16x16 C tiles, K-dim = hd = 64 ----
    v8f s0 = wmma_f16(aq0, bk00, zero8());
    s0 = wmma_f16(aq1, bk01, s0);
    v8f s1 = wmma_f16(aq0, bk10, zero8());
    s1 = wmma_f16(aq1, bk11, s1);

    // ---- causal mask + online softmax ----
    const int key0 = k0 + cl, key1 = k0 + 16 + cl;
#pragma unroll
    for (int r = 0; r < 8; ++r) {
      const int qrow = q0 + r + 8 * hg;
      float x0 = (key0 <= qrow) ? s0[r] : -1e30f;
      float x1 = (key1 <= qrow) ? s1[r] : -1e30f;
      float mx = fmaxf(x0, x1);
      mx = fmaxf(mx, __shfl_xor(mx, 1, 32));
      mx = fmaxf(mx, __shfl_xor(mx, 2, 32));
      mx = fmaxf(mx, __shfl_xor(mx, 4, 32));
      mx = fmaxf(mx, __shfl_xor(mx, 8, 32));   // row max within half-wave
      float mnew = fmaxf(mrun[r], mx);
      float alpha = __expf(mrun[r] - mnew);
      float p0 = __expf(x0 - mnew);
      float p1 = __expf(x1 - mnew);
      float rs = p0 + p1;
      rs += __shfl_xor(rs, 1, 32);
      rs += __shfl_xor(rs, 2, 32);
      rs += __shfl_xor(rs, 4, 32);
      rs += __shfl_xor(rs, 8, 32);
      mrun[r] = mnew;
      lrun[r] = lrun[r] * alpha + rs;
      o0[r] *= alpha; o1[r] *= alpha; o2[r] *= alpha; o3[r] *= alpha;
      const int prow = r + 8 * hg;
      Pw[prow * 40 + cl] = (_Float16)p0;        // ds_store_b16
      Pw[prow * 40 + 16 + cl] = (_Float16)p1;
    }

    // ---- P (16x32) as A-operand from LDS; V tiles as B-operand ----
    v16h ap = load_op16(Pw + cl * 40 + kbase);  // ds_load_b128 x2
    const _Float16* vb = vT + ((size_t)bh * HD_ + cl) * S_ + k0 + kbase;
    __builtin_prefetch(vb + 32, 0, 3);          // next V chunk
    o0 = wmma_f16(ap, load_op16(vb), o0);
    o1 = wmma_f16(ap, load_op16(vb + 16 * S_), o1);
    o2 = wmma_f16(ap, load_op16(vb + 32 * S_), o2);
    o3 = wmma_f16(ap, load_op16(vb + 48 * S_), o3);
  }

  // ---- normalize, write ctx as (b, s, h*64+d) f16 for the final GEMM ----
  const int b = bh >> 4;
  const int h = bh & 15;
#pragma unroll
  for (int r = 0; r < 8; ++r) {
    const int qrow = q0 + r + 8 * hg;
    float inv = 1.0f / lrun[r];
    size_t base = ((size_t)b * S_ + qrow) * DOUT + h * HD_ + cl;
    ctx[base]      = (_Float16)(o0[r] * inv);
    ctx[base + 16] = (_Float16)(o1[r] * inv);
    ctx[base + 32] = (_Float16)(o2[r] * inv);
    ctx[base + 48] = (_Float16)(o3[r] * inv);
  }
}

// ---------------------------------------------------------------------------
// launcher
// ---------------------------------------------------------------------------
extern "C" void kernel_launch(void* const* d_in, const int* in_sizes, int n_in,
                              void* d_out, int out_size, void* d_ws, size_t ws_size,
                              hipStream_t stream) {
  (void)in_sizes; (void)n_in; (void)out_size; (void)ws_size;
  const float* x  = (const float*)d_in[0];
  const float* Wq = (const float*)d_in[1];
  const float* Wk = (const float*)d_in[2];
  const float* Wv = (const float*)d_in[3];
  const float* Wo = (const float*)d_in[4];
  float* out = (float*)d_out;

  char* ws = (char*)d_ws;
  size_t off = 0;
  _Float16* xh  = (_Float16*)(ws + off); off += (size_t)B_ * S_ * DIN * 2;
  _Float16* WqT = (_Float16*)(ws + off); off += (size_t)DIN * DOUT * 2;
  _Float16* WkT = (_Float16*)(ws + off); off += (size_t)DIN * DOUT * 2;
  _Float16* WvT = (_Float16*)(ws + off); off += (size_t)DIN * DOUT * 2;
  _Float16* WoT = (_Float16*)(ws + off); off += (size_t)DIN * DOUT * 2;
  _Float16* qhb = (_Float16*)(ws + off); off += (size_t)B_ * S_ * DOUT * 2;
  _Float16* khb = (_Float16*)(ws + off); off += (size_t)B_ * S_ * DOUT * 2;
  _Float16* vTb = (_Float16*)(ws + off); off += (size_t)B_ * S_ * DOUT * 2;
  _Float16* ctxh = (_Float16*)(ws + off); off += (size_t)B_ * S_ * DOUT * 2;

  mha_convert<<<1024, 256, 0, stream>>>(x, Wq, Wk, Wv, Wo, xh, WqT, WkT, WvT, WoT);

  dim3 g((B_ * S_) / 64, DOUT / 256), blk(256);
  mha_gemm_wmma<<<g, blk, 0, stream>>>(xh, WqT, qhb, B_ * S_, DOUT, DIN, 0);
  mha_gemm_wmma<<<g, blk, 0, stream>>>(xh, WkT, khb, B_ * S_, DOUT, DIN, 1);
  mha_gemm_wmma<<<g, blk, 0, stream>>>(xh, WvT, vTb, B_ * S_, DOUT, DIN, 2);

  mha_flash_wmma<<<B_ * H_ * (S_ / 64), 128, 0, stream>>>(qhb, khb, vTb, ctxh);

  mha_gemm_wmma<<<g, blk, 0, stream>>>(ctxh, WoT, out, B_ * S_, DOUT, DIN, 3);
}
